// CenternetDeconv1_82497731822220
// MI455X (gfx1250) — matrix operand order repaired and checked by reading, not compiled, not run on hardware
//
#include <hip/hip_runtime.h>

typedef __attribute__((ext_vector_type(16))) __bf16 bf16x16;
typedef __attribute__((ext_vector_type(8)))  __bf16 bf16x8;
typedef __attribute__((ext_vector_type(8)))  float  f32x8;

// ---------------------------------------------------------------------------
// Async global->LDS copy of one 16-byte chunk per lane (ASYNCcnt-tracked).
// dst must be a pointer into __shared__; low 32 bits of the generic address
// are the LDS offset (ISA 10.2 aperture rule).
// ---------------------------------------------------------------------------
__device__ __forceinline__ void async_copy_b128(const __bf16* gsrc, __bf16* ldst)
{
    unsigned loff = (unsigned)(uintptr_t)ldst;
    asm volatile("global_load_async_to_lds_b128 %0, %1, off"
                 :: "v"(loff), "v"(gsrc) : "memory");
}

// ---------------------------------------------------------------------------
// WMMA GEMM:  out = epilogue( A[M,K] * B[K,Np] )
//   A  : bf16 [M][K] row-major (weights, M multiple of 16, K multiple of 64)
//   Bm : bf16 [Np][K] row-major  (== B stored column-major / pixel-major)
// Block = 128 threads (4 waves), all waves share one m-tile whose A slice is
// double-buffered in LDS via global_load_async_to_lds_b128; each wave computes
// 4 n-tiles (4 f32x8 accumulators) per staged 16x64 A tile.
// modes:
//   0: f32 pixel-major  out[n*ldo + co + m] = acc + (m<nbias ? bias[m] : 0)
//   1: bf16 pixel-major out[n*ldo + co + m] = relu((acc+bias[m])*scale[m]+shift[m])
//   2: f32 NCHW         out[((b*Cn+m)*HW)+p] = relu((acc+bias[m])*scale[m]+shift[m])
// ---------------------------------------------------------------------------
#define APITCH 72   // bf16 entries per staged A row: 144B = 36 dwords -> no bank conflicts
__global__ __launch_bounds__(128)
void gemm_wmma_bf16(const __bf16* __restrict__ A,
                    const __bf16* __restrict__ Bm,
                    int M, int K, int Np,
                    const float* __restrict__ bias, int nbias,
                    const float* __restrict__ scale,
                    const float* __restrict__ shift,
                    void* __restrict__ outp, int ldo, int co,
                    int mode, int Cn, int HW)
{
    __shared__ __bf16 smA[2][16 * APITCH];

    const int tid  = threadIdx.x;
    const int lane = tid & 31;
    const int wave = tid >> 5;
    const int half = lane >> 4;     // K-half selector (wave32 WMMA layout)
    const int l16  = lane & 15;
    const int m0   = blockIdx.y * 16;
    const int nt0  = blockIdx.x * 16 + wave * 4;   // first of 4 n-tiles for this wave

    // B row pointers (clamped so inactive waves still compute with full EXEC)
    const __bf16* brow[4];
#pragma unroll
    for (int t = 0; t < 4; ++t) {
        int n = (nt0 + t) * 16 + l16;
        if (n >= Np) n = Np - 1;
        brow[t] = Bm + (size_t)n * K;
    }

    // A staging: 128 threads x 16B = 2KB = one 16x64 bf16 tile per step
    const int arow_s = tid >> 3;          // 0..15
    const int acol_s = (tid & 7) * 8;     // 0..56
    const __bf16* asrc = A + (size_t)(m0 + arow_s) * K + acol_s;
    __bf16* adst[2] = { &smA[0][arow_s * APITCH + acol_s],
                        &smA[1][arow_s * APITCH + acol_s] };

    f32x8 acc[4] = {};

    async_copy_b128(asrc, adst[0]);                // prefill buffer 0
    int cur = 0;
    for (int kb = 0; kb < K; kb += 64) {
        if (kb + 64 < K) {                         // issue next tile, retire current
            async_copy_b128(asrc + kb + 64, adst[cur ^ 1]);
            asm volatile("s_wait_asynccnt 0x1" ::: "memory");
        } else {
            asm volatile("s_wait_asynccnt 0x0" ::: "memory");
        }
        __syncthreads();                           // all waves' staged data visible

        const __bf16* ab = &smA[cur][l16 * APITCH];
#pragma unroll
        for (int ks = 0; ks < 2; ++ks) {
            // A fragment from LDS: K = half*8+[0..7] and 16+half*8+[0..7]
            bf16x8 alo = *(const bf16x8*)(ab + ks * 32 + half * 8);
            bf16x8 ahi = *(const bf16x8*)(ab + ks * 32 + 16 + half * 8);
            bf16x16 a_frag = __builtin_shufflevector(alo, ahi,
                             0,1,2,3,4,5,6,7,8,9,10,11,12,13,14,15);
            int ko = kb + ks * 32;
#pragma unroll
            for (int t = 0; t < 4; ++t) {
                // B fragment: K = half*8+[0..7] and 16+half*8+[0..7] (matches A)
                bf16x16 b_frag = __builtin_shufflevector(*(const bf16x8*)(brow[t] + ko + half * 8), *(const bf16x8*)(brow[t] + ko + 16 + half * 8), 0,1,2,3,4,5,6,7,8,9,10,11,12,13,14,15);
                acc[t] = __builtin_amdgcn_wmma_f32_16x16x32_bf16(
                             false, a_frag, false, b_frag, (short)0, acc[t],
                             false, false);
            }
        }
        if (kb + 512 < K) {                        // keep B stream warm in-cache
#pragma unroll
            for (int t = 0; t < 4; ++t)
                __builtin_prefetch(brow[t] + kb + 512, 0, 3);
        }
        __syncthreads();                           // protect buffer for overwrite
        cur ^= 1;
    }

    // D layout: lane holds column n, rows m0 + half*8 + r, r=0..7 (contiguous)
    const int mb = m0 + half * 8;
#pragma unroll
    for (int t = 0; t < 4; ++t) {
        if ((nt0 + t) * 16 >= Np) break;           // wave-uniform
        const int n = (nt0 + t) * 16 + l16;
        if (mode == 0) {
            float* o = (float*)outp + (size_t)n * ldo + co + mb;
            f32x8 v;
#pragma unroll
            for (int r = 0; r < 8; ++r) {
                int m = mb + r;
                v[r] = acc[t][r] + ((bias && m < nbias) ? bias[m] : 0.f);
            }
            *(f32x8*)o = v;
        } else if (mode == 1) {
            __bf16* o = (__bf16*)outp + (size_t)n * ldo + co + mb;
            bf16x8 v;
#pragma unroll
            for (int r = 0; r < 8; ++r) {
                int m = mb + r;
                float x = (acc[t][r] + bias[m]) * scale[m] + shift[m];
                v[r] = (__bf16)(x > 0.f ? x : 0.f);
            }
            *(bf16x8*)o = v;
        } else { // mode 2: f32 NCHW
            int b = n / HW, rem = n % HW;
            float* o = (float*)outp;
#pragma unroll
            for (int r = 0; r < 8; ++r) {
                int m = mb + r;
                float x = (acc[t][r] + bias[m]) * scale[m] + shift[m];
                o[((size_t)b * Cn + m) * HW + rem] = x > 0.f ? x : 0.f;
            }
        }
    }
}

// ---------------------------------------------------------------------------
// Support kernels (grid-stride, plain VALU)
// ---------------------------------------------------------------------------
__global__ void maxpool_pxc(const float* __restrict__ in, __bf16* __restrict__ out,
                            int B, int C, int H, int W)
{
    int Ho = H / 2, Wo = W / 2;
    long long total = (long long)B * Ho * Wo * C;
    for (long long i = (long long)blockIdx.x * blockDim.x + threadIdx.x; i < total;
         i += (long long)gridDim.x * blockDim.x) {
        int c = (int)(i % C); long long p = i / C;
        int x = (int)(p % Wo); int y = (int)((p / Wo) % Ho); int b = (int)(p / ((long long)Wo * Ho));
        const float* s = in + ((size_t)(b * C + c) * H + 2 * y) * W + 2 * x;
        float m = fmaxf(fmaxf(s[0], s[1]), fmaxf(s[W], s[W + 1]));
        out[(size_t)p * C + c] = (__bf16)m;
    }
}

__global__ void nchw_to_pxc(const float* __restrict__ in, __bf16* __restrict__ out,
                            int B, int C, int H, int W, int ldo, int co)
{
    long long total = (long long)B * H * W * C;
    for (long long i = (long long)blockIdx.x * blockDim.x + threadIdx.x; i < total;
         i += (long long)gridDim.x * blockDim.x) {
        int c = (int)(i % C); long long p = i / C;
        int x = (int)(p % W); int y = (int)((p / W) % H); int b = (int)(p / ((long long)W * H));
        out[(size_t)p * ldo + co + c] =
            (__bf16)in[((size_t)(b * C + c) * H + y) * W + x];
    }
}

// dw [Cout][Cin][3][3] f32 -> [Mpad][kk*Cin + c] bf16 (zero-padded rows)
__global__ void repack_w3x3(const float* __restrict__ w, __bf16* __restrict__ o,
                            int Cout, int Mpad, int Cin)
{
    long long total = (long long)Mpad * 9 * Cin;
    for (long long i = (long long)blockIdx.x * blockDim.x + threadIdx.x; i < total;
         i += (long long)gridDim.x * blockDim.x) {
        int c = (int)(i % Cin); long long t = i / Cin;
        int kk = (int)(t % 9); int m = (int)(t / 9);
        float v = (m < Cout) ? w[((size_t)m * Cin + c) * 9 + kk] : 0.f;
        o[i] = (__bf16)v;
    }
}

// up_w [Cin][Cout][4][4] f32 -> [(oc*16+kk)][ci] bf16
__global__ void repack_up(const float* __restrict__ w, __bf16* __restrict__ o,
                          int Cin, int Cout)
{
    long long total = (long long)Cout * 16 * Cin;
    for (long long i = (long long)blockIdx.x * blockDim.x + threadIdx.x; i < total;
         i += (long long)gridDim.x * blockDim.x) {
        int ci = (int)(i % Cin); long long t = i / Cin;
        int kk = (int)(t % 16); int oc = (int)(t / 16);
        o[i] = (__bf16)w[((size_t)ci * Cout + oc) * 16 + kk];
    }
}

__global__ void bnprep(const float* __restrict__ g, const float* __restrict__ b,
                       const float* __restrict__ m, const float* __restrict__ v,
                       float* __restrict__ scale, float* __restrict__ shift, int C)
{
    int i = blockIdx.x * blockDim.x + threadIdx.x;
    if (i < C) {
        float s = g[i] * rsqrtf(v[i] + 1e-5f);
        scale[i] = s;
        shift[i] = b[i] - m[i] * s;
    }
}

// regular im2col (3x3, pad 1) from pixel-major bf16 activations
__global__ void im2col_reg(const __bf16* __restrict__ act, __bf16* __restrict__ col,
                           int B, int C, int H, int W)
{
    long long total = (long long)B * H * W * 9 * C;
    for (long long i = (long long)blockIdx.x * blockDim.x + threadIdx.x; i < total;
         i += (long long)gridDim.x * blockDim.x) {
        int c = (int)(i % C); long long t = i / C;
        int kk = (int)(t % 9); long long p = t / 9;
        int x = (int)(p % W); int y = (int)((p / W) % H); int b = (int)(p / ((long long)W * H));
        int yy = y + kk / 3 - 1, xx = x + kk % 3 - 1;
        __bf16 v = (__bf16)0.f;
        if (yy >= 0 && yy < H && xx >= 0 && xx < W)
            v = act[(((size_t)b * H + yy) * W + xx) * C + c];
        col[(size_t)p * (9 * C) + (size_t)kk * C + c] = v;
    }
}

__device__ __forceinline__ float dcn_tap(const __bf16* act, int b, int C, int H, int W,
                                         int c, float yi, float xi, float wgt)
{
    if (!(yi >= 0.f && yi <= (float)(H - 1) && xi >= 0.f && xi <= (float)(W - 1)))
        return 0.f;
    int yc = (int)yi, xc = (int)xi;
    return wgt * (float)act[(((size_t)b * H + yc) * W + xc) * C + c];
}

// bilinear deformable sampling + sigmoid modulation -> val[p][kk*C+c] bf16
__global__ void deform_gather(const __bf16* __restrict__ act, const float* __restrict__ om,
                              __bf16* __restrict__ val, int B, int C, int H, int W)
{
    long long total = (long long)B * H * W * 9 * C;
    for (long long i = (long long)blockIdx.x * blockDim.x + threadIdx.x; i < total;
         i += (long long)gridDim.x * blockDim.x) {
        int c = (int)(i % C); long long t = i / C;
        int kk = (int)(t % 9); long long p = t / 9;
        int x = (int)(p % W); int y = (int)((p / W) % H); int b = (int)(p / ((long long)W * H));
        const float* o = om + (size_t)p * 32;
        float dy = o[2 * kk], dx = o[2 * kk + 1];
        float mm = 1.f / (1.f + __expf(-o[18 + kk]));
        float py = (float)(y + kk / 3 - 1) + dy;
        float px = (float)(x + kk % 3 - 1) + dx;
        float y0 = floorf(py), x0 = floorf(px);
        float wy = py - y0, wx = px - x0;
        float s = dcn_tap(act, b, C, H, W, c, y0,       x0,       (1.f - wy) * (1.f - wx))
                + dcn_tap(act, b, C, H, W, c, y0,       x0 + 1.f, (1.f - wy) * wx)
                + dcn_tap(act, b, C, H, W, c, y0 + 1.f, x0,       wy * (1.f - wx))
                + dcn_tap(act, b, C, H, W, c, y0 + 1.f, x0 + 1.f, wy * wx);
        val[(size_t)p * (9 * C) + (size_t)kk * C + c] = (__bf16)(s * mm);
    }
}

// col2im for ConvTranspose2d(4, stride 2, pad 1): pure gather of 4 Z entries,
// fused BN+ReLU, bf16 pixel-major output (into a concat-buffer column slice)
__global__ void ctrans_out(const float* __restrict__ Z, __bf16* __restrict__ out,
                           const float* __restrict__ scale, const float* __restrict__ shift,
                           int B, int Cout, int H, int W, int ldo, int co)
{
    int H2 = 2 * H, W2 = 2 * W;
    long long total = (long long)B * H2 * W2 * Cout;
    for (long long i = (long long)blockIdx.x * blockDim.x + threadIdx.x; i < total;
         i += (long long)gridDim.x * blockDim.x) {
        int oc = (int)(i % Cout); long long p = i / Cout;
        int ox = (int)(p % W2); int oy = (int)((p / W2) % H2); int b = (int)(p / ((long long)W2 * H2));
        int ky0 = (oy + 1) & 1, kx0 = (ox + 1) & 1;
        float s = 0.f;
#pragma unroll
        for (int a = 0; a < 2; ++a)
#pragma unroll
            for (int d = 0; d < 2; ++d) {
                int ky = ky0 + 2 * a, kx = kx0 + 2 * d;
                int iy = (oy + 1 - ky) >> 1, ix = (ox + 1 - kx) >> 1;
                if (iy >= 0 && iy < H && ix >= 0 && ix < W)
                    s += Z[(((size_t)b * H + iy) * W + ix) * ((size_t)Cout * 16)
                           + (size_t)oc * 16 + ky * 4 + kx];
            }
        float v = s * scale[oc] + shift[oc];
        out[(size_t)p * ldo + co + oc] = (__bf16)(v > 0.f ? v : 0.f);
    }
}

// ---------------------------------------------------------------------------
// Host orchestration
// ---------------------------------------------------------------------------
struct BNp  { const float *g, *b, *m, *v; };
struct DCNp { const float *db, *dw, *ob, *ow; };
struct DLp  { BNp bn1, bn2; DCNp dcn; const float* up; };

static inline unsigned nb(long long t)
{
    long long b = (t + 255) / 256;
    if (b > 1048576) b = 1048576;
    if (b < 1) b = 1;
    return (unsigned)b;
}

extern "C" void kernel_launch(void* const* d_in, const int* in_sizes, int n_in,
                              void* d_out, int out_size, void* d_ws, size_t ws_size,
                              hipStream_t stream)
{
    (void)in_sizes; (void)n_in; (void)out_size; (void)ws_size;

    const float* res2 = (const float*)d_in[0]; // [2, 64,128,128]
    const float* res3 = (const float*)d_in[1]; // [2,128, 64, 64]
    const float* res4 = (const float*)d_in[2]; // [2,256, 32, 32]
    const float* res5 = (const float*)d_in[3]; // [2,512, 16, 16]

    auto F = [&](int i) { return (const float*)d_in[i]; };
    // JAX pytree flatten: top-level insertion order given; nested dicts by sorted key.
    BNp  c4bn1{F(4), F(5), F(6), F(7)},  c4bn2{F(8), F(9), F(10), F(11)};
    DCNp c4d1{F(12), F(13), F(14), F(15)}, c4d2{F(16), F(17), F(18), F(19)};
    auto getDL = [&](int base) {
        DLp d;
        d.bn1 = {F(base + 0), F(base + 1), F(base + 2),  F(base + 3)};
        d.bn2 = {F(base + 4), F(base + 5), F(base + 6),  F(base + 7)};
        d.dcn = {F(base + 8), F(base + 9), F(base + 10), F(base + 11)};
        d.up  =  F(base + 12);
        return d;
    };
    DLp dv0 = getDL(20), dv1 = getDL(33), dv2 = getDL(46), dv3 = getDL(59);

    // ---- workspace carve (all 256B aligned) ----
    char* wp = (char*)d_ws;
    auto carve = [&](size_t bytes) { void* r = wp; wp += (bytes + 255) & ~(size_t)255; return r; };
    __bf16* COL   = (__bf16*)carve(37748736ull * 2); // im2col / deform-val (<= L2!)
    float*  Zbuf  = (float*) carve( 8388608ull * 4); // transpose-conv col
    float*  OM    = (float*) carve( 1048576ull * 4); // offset/mask conv out [p][32]
    __bf16* WMAIN = (__bf16*)carve( 2359296ull * 2);
    __bf16* WOFF  = (__bf16*)carve(  294912ull * 2);
    __bf16* WUP   = (__bf16*)carve( 4194304ull * 2);
    float*  BNS   = (float*) carve(1024 * 4);
    float*  BNH   = (float*) carve(1024 * 4);
    __bf16* X0    = (__bf16*)carve(  65536ull * 2);  // [128][512]
    __bf16* CAT1  = (__bf16*)carve( 524288ull * 2);  // [512][1024]
    __bf16* CAT2  = (__bf16*)carve(1048576ull * 2);  // [2048][512]
    __bf16* CAT3  = (__bf16*)carve(2097152ull * 2);  // [8192][256]
    __bf16* CAT4  = (__bf16*)carve(4194304ull * 2);  // [32768][128]
    __bf16* MID   = (__bf16*)carve(2097152ull * 2);  // [32768][64]
    __bf16* MDOUT = (__bf16*)carve( 524288ull * 2);  // mdcn out before upsample

    auto gemm = [&](const __bf16* A, const __bf16* Bm, int M, int K, int Np,
                    const float* bias, int nbias, const float* sc, const float* sh,
                    void* out, int ldo, int co, int mode, int Cn, int HW) {
        dim3 g((Np + 255) / 256, M / 16);
        gemm_wmma_bf16<<<g, dim3(128), 0, stream>>>(A, Bm, M, K, Np, bias, nbias,
                                                    sc, sh, out, ldo, co, mode, Cn, HW);
    };

    // modulated deformable conv: x (bf16 pxc [p][C]) -> out via mode 1 (bf16) or 2 (f32 NCHW)
    auto run_mdcn = [&](const __bf16* x, int Bn, int C, int H, int W, int Cout,
                        const DCNp& d, const BNp& bn, void* out, int ldo,
                        int mode, int HW) {
        int Np = Bn * H * W, K = 9 * C;
        repack_w3x3<<<nb((long long)Cout * K), 256, 0, stream>>>(d.dw, WMAIN, Cout, Cout, C);
        repack_w3x3<<<nb(32LL * K),            256, 0, stream>>>(d.ow, WOFF, 27, 32, C);
        im2col_reg<<<nb((long long)Np * K),    256, 0, stream>>>(x, COL, Bn, C, H, W);
        gemm(WOFF, COL, 32, K, Np, d.ob, 27, nullptr, nullptr, OM, 32, 0, 0, 0, 0);
        deform_gather<<<nb((long long)Np * K), 256, 0, stream>>>(x, OM, COL, Bn, C, H, W);
        bnprep<<<(Cout + 255) / 256, 256, 0, stream>>>(bn.g, bn.b, bn.m, bn.v, BNS, BNH, Cout);
        gemm(WMAIN, COL, Cout, K, Np, d.db, Cout, BNS, BNH, out, ldo, 0, mode, Cout, HW);
    };

    // transpose conv (4,2,1) + BN/ReLU into concat buffer columns [0,Cc)
    auto run_up = [&](const __bf16* xin, int Bn, int Cc, int H, int W,
                      const float* upw, const BNp& bn, __bf16* outcat, int ldo) {
        repack_up<<<nb((long long)Cc * 16 * Cc), 256, 0, stream>>>(upw, WUP, Cc, Cc);
        int Np = Bn * H * W, M = Cc * 16;
        gemm(WUP, xin, M, Cc, Np, nullptr, 0, nullptr, nullptr, Zbuf, M, 0, 0, 0, 0);
        bnprep<<<(Cc + 255) / 256, 256, 0, stream>>>(bn.g, bn.b, bn.m, bn.v, BNS, BNH, Cc);
        ctrans_out<<<nb((long long)Bn * 4 * H * W * Cc), 256, 0, stream>>>(
            Zbuf, outcat, BNS, BNH, Bn, Cc, H, W, ldo, 0);
    };

    // ---------------- pipeline ----------------
    maxpool_pxc<<<nb(2LL * 8 * 8 * 512), 256, 0, stream>>>(res5, X0, 2, 512, 16, 16);

    // deconv0: mdcn 512->512 @8x8, upsample -> 16x16 into CAT1[:, :512]
    run_mdcn(X0, 2, 512, 8, 8, 512, dv0.dcn, dv0.bn1, MDOUT, 512, 1, 0);
    run_up(MDOUT, 2, 512, 8, 8, dv0.up, dv0.bn2, CAT1, 1024);
    nchw_to_pxc<<<nb(2LL * 16 * 16 * 512), 256, 0, stream>>>(res5, CAT1, 2, 512, 16, 16, 1024, 512);

    // deconv1: mdcn 1024->256 @16x16, upsample -> 32x32 into CAT2[:, :256]
    run_mdcn(CAT1, 2, 1024, 16, 16, 256, dv1.dcn, dv1.bn1, MDOUT, 256, 1, 0);
    run_up(MDOUT, 2, 256, 16, 16, dv1.up, dv1.bn2, CAT2, 512);
    nchw_to_pxc<<<nb(2LL * 32 * 32 * 256), 256, 0, stream>>>(res4, CAT2, 2, 256, 32, 32, 512, 256);

    // deconv2: mdcn 512->128 @32x32, upsample -> 64x64 into CAT3[:, :128]
    run_mdcn(CAT2, 2, 512, 32, 32, 128, dv2.dcn, dv2.bn1, MDOUT, 128, 1, 0);
    run_up(MDOUT, 2, 128, 32, 32, dv2.up, dv2.bn2, CAT3, 256);
    nchw_to_pxc<<<nb(2LL * 64 * 64 * 128), 256, 0, stream>>>(res3, CAT3, 2, 128, 64, 64, 256, 128);

    // deconv3: mdcn 256->64 @64x64, upsample -> 128x128 into CAT4[:, :64]
    run_mdcn(CAT3, 2, 256, 64, 64, 64, dv3.dcn, dv3.bn1, MDOUT, 64, 1, 0);
    run_up(MDOUT, 2, 64, 64, 64, dv3.up, dv3.bn2, CAT4, 128);
    nchw_to_pxc<<<nb(2LL * 128 * 128 * 64), 256, 0, stream>>>(res2, CAT4, 2, 64, 128, 128, 128, 64);

    // conv4: mdcn 128->64, then mdcn 64->64 with f32 NCHW output
    run_mdcn(CAT4, 2, 128, 128, 128, 64, c4d1, c4bn1, MID, 64, 1, 0);
    run_mdcn(MID, 2, 64, 128, 128, 64, c4d2, c4bn2, d_out, 0, 2, 128 * 128);
}
